// CustomJacobiLayer_20358144983511
// MI455X (gfx1250) — matrix operand order repaired and checked by reading, not compiled, not run on hardware
//
#include <hip/hip_runtime.h>
#include <hip/hip_fp16.h>

typedef __attribute__((ext_vector_type(16))) _Float16 v16h;
typedef __attribute__((ext_vector_type(8)))  _Float16 v8h;
typedef __attribute__((ext_vector_type(8)))  float    v8f;

#define JORDER 8
#define KP     (JORDER + 1)   // 9
#define IN     512
#define OUT    512
#define BATCH  4096
#define KTOT   (IN * KP)      // 4608, divisible by 32

// WGP-scope speculative prefetch (scope omitted -> 0 = WGP: pulls into ALL
// cache levels incl. the WGP cache). Uses the instruction's 24-bit immediate
// offset so no new address VGPRs are written (avoids WMMA->VALU WAR hazard
// nops on the fragment registers).
#define PREFETCH_WGP(ptr, imm)                                            \
  asm volatile("global_prefetch_b8 %0, off offset:%1"                    \
               :: "v"((const void*)(ptr)), "n"(imm))

// ---------------------------------------------------------------------------
// Kernel 1: W2[o*KTOT + i*9 + k] = coeff[o,i,k] * weights[o,i]   (f32 -> f16)
// ---------------------------------------------------------------------------
__global__ void fuse_wc(const float* __restrict__ W, const float* __restrict__ C,
                        _Float16* __restrict__ W2) {
  int idx = blockIdx.x * blockDim.x + threadIdx.x;     // over OUT*IN
  if (idx >= OUT * IN) return;
  float w = W[idx];
  const float* c = C + (size_t)idx * KP;
  _Float16* o = W2 + (size_t)idx * KP;
  #pragma unroll
  for (int k = 0; k < KP; ++k) o[k] = (_Float16)(c[k] * w);
}

// ---------------------------------------------------------------------------
// Kernel 2: Jf[b*KTOT + i*9 + k] = Jacobi_k(tanh(x[b,i]))   (f16)
// alpha = beta = 1; recurrence constants constant-fold (unrolled, literal i).
// ---------------------------------------------------------------------------
__global__ void jacobi_k(const float* __restrict__ X, _Float16* __restrict__ Jf) {
  int idx = blockIdx.x * blockDim.x + threadIdx.x;     // over BATCH*IN
  if (idx >= BATCH * IN) return;
  const float A = 1.0f, Bp = 1.0f;
  float t = tanhf(X[idx]);
  _Float16* o = Jf + (size_t)idx * KP;
  float pm2 = 1.0f;
  float pm1 = 0.5f * (A + Bp + 2.0f) * t - 0.5f * (A - Bp);
  o[0] = (_Float16)pm2;
  o[1] = (_Float16)pm1;
  #pragma unroll
  for (int i = 2; i <= JORDER; ++i) {
    float fi = (float)i;
    float ab = A + Bp;
    float k1 = (2.0f * fi + ab) * (2.0f * fi + ab - 1.0f) / (2.0f * fi * (fi + ab));
    float k2 = (2.0f * fi + ab - 1.0f) * (A * A - Bp * Bp) /
               (2.0f * fi * (fi + ab) * (2.0f * fi + ab - 2.0f));
    float k3 = (fi + A - 1.0f) * (fi + Bp - 1.0f) * (2.0f * fi + ab) /
               (fi * (fi + ab) * (2.0f * fi + ab - 2.0f));
    float p = (k1 * t + k2) * pm1 - k3 * pm2;
    o[i] = (_Float16)p;
    pm2 = pm1; pm1 = p;
  }
}

// ---------------------------------------------------------------------------
// Kernel 3: out[4096,512] = Jf[4096,4608] * W2^T, v_wmma_f32_16x16x32_f16.
// 2x2 register-blocked: each wave computes a 32x32 output tile with 4
// accumulators; per K-step loads 2 A-frags + 2 B-frags for 4 WMMAs.
//
// Per-lane fragment layout (CDNA5 ISA 7.12.2, 16-bit, wave32):
//   A 16x32 (MxK): lane<16 -> row=lane,     K = {k0+0..7,  k0+16..23}
//                  lane>=16 -> row=lane-16, K = {k0+8..15, k0+24..31}
//   B 32x16 (KxN): lane<16 -> col=lane,     K = {k0+0..7,  k0+16..23}
//                  lane>=16 -> col=lane-16, K = {k0+8..15, k0+24..31}
//   D 16x16 f32:  VGPR r -> M = r + (lane<16 ? 0 : 8), N = lane&15
// ---------------------------------------------------------------------------
__global__ void __launch_bounds__(256) gemm_wmma(const _Float16* __restrict__ Jf,
                                                 const _Float16* __restrict__ W2,
                                                 float* __restrict__ Out) {
  const int lane = threadIdx.x & 31;
  const int wave = threadIdx.x >> 5;
  const int tilesN = OUT / 32;                 // 16
  int tile = blockIdx.x * 8 + wave;            // 2048 wave-tiles total
  int tm = tile / tilesN;                      // 0..127  (rows of 32)
  int tn = tile % tilesN;                      // 0..15   (cols of 32)

  const int half = (lane < 16) ? 1 : 0;        // 1 = lower half-wave
  const int rowA = tm * 32 + (lane & 15);
  const int colB = tn * 32 + (lane & 15);

  const _Float16* a0 = Jf + (size_t)rowA * KTOT + (half ? 0 : 8);
  const _Float16* a1 = a0 + (size_t)16 * KTOT;
  const _Float16* b0 = W2 + (size_t)colB * KTOT + (half ? 0 : 8);
  const _Float16* b1 = b0 + (size_t)16 * KTOT;

  v8f acc00 = {}, acc01 = {}, acc10 = {}, acc11 = {};

  // Prefetch distances: +256 B (~4 K-steps ahead, one line per unrolled body
  // per stream); FAR adds the a1/b1 stream distance (16*KTOT*2 bytes).
  constexpr int PF_NEAR = 256;
  constexpr int PF_FAR  = 16 * KTOT * 2 + 256;          // 147712 (< 2^23)

  #pragma unroll 2
  for (int k0 = 0; k0 < KTOT; k0 += 32) {
    union { v16h v; v8h h[2]; } fa0, fa1, fb0, fb1;
    fa0.h[0] = *(const v8h*)(a0 + k0);
    fa0.h[1] = *(const v8h*)(a0 + k0 + 16);
    fa1.h[0] = *(const v8h*)(a1 + k0);
    fa1.h[1] = *(const v8h*)(a1 + k0 + 16);
    fb0.h[0] = *(const v8h*)(b0 + k0);
    fb0.h[1] = *(const v8h*)(b0 + k0 + 16);
    fb1.h[0] = *(const v8h*)(b1 + k0);
    fb1.h[1] = *(const v8h*)(b1 + k0 + 16);

    // One line-prefetch per stream per 64-byte-pair body: only the first
    // copy of the unrolled pair emits them ((k0 & 32) folds at compile time),
    // so the operand is always the live loop-base register -> no extra adds,
    // no WMMA->VALU hazard padding.
    if ((k0 & 32) == 0) {
      PREFETCH_WGP(a0 + k0, PF_NEAR);
      PREFETCH_WGP(a0 + k0, PF_FAR);
      PREFETCH_WGP(b0 + k0, PF_NEAR);
      PREFETCH_WGP(b0 + k0, PF_FAR);
    }

    acc00 = __builtin_amdgcn_wmma_f32_16x16x32_f16(false, fa0.v, false, fb0.v,
                                                   (short)0, acc00, false, false);
    acc01 = __builtin_amdgcn_wmma_f32_16x16x32_f16(false, fa0.v, false, fb1.v,
                                                   (short)0, acc01, false, false);
    acc10 = __builtin_amdgcn_wmma_f32_16x16x32_f16(false, fa1.v, false, fb0.v,
                                                   (short)0, acc10, false, false);
    acc11 = __builtin_amdgcn_wmma_f32_16x16x32_f16(false, fa1.v, false, fb1.v,
                                                   (short)0, acc11, false, false);
  }

  // Store the four 16x16 sub-tiles.
  const int mofs = half ? 0 : 8;
  float* obase = Out + (size_t)(tm * 32) * OUT + tn * 32 + (lane & 15);
  #pragma unroll
  for (int r = 0; r < 8; ++r) {
    size_t rr = (size_t)(r + mofs) * OUT;
    obase[rr]                          = acc00[r];  // rows +0..15,  cols +0..15
    obase[rr + 16]                     = acc01[r];  // rows +0..15,  cols +16..31
    obase[rr + (size_t)16 * OUT]       = acc10[r];  // rows +16..31, cols +0..15
    obase[rr + (size_t)16 * OUT + 16]  = acc11[r];  // rows +16..31, cols +16..31
  }
}

// ---------------------------------------------------------------------------
extern "C" void kernel_launch(void* const* d_in, const int* in_sizes, int n_in,
                              void* d_out, int out_size, void* d_ws, size_t ws_size,
                              hipStream_t stream) {
  const float* x = (const float*)d_in[0];   // [BATCH, IN]
  const float* w = (const float*)d_in[1];   // [OUT, IN]
  const float* c = (const float*)d_in[2];   // [OUT, IN, KP]
  float* out = (float*)d_out;               // [BATCH, OUT]

  // Workspace layout: W2 (f16, OUT*KTOT) | Jf (f16, BATCH*KTOT)
  _Float16* W2 = (_Float16*)d_ws;
  size_t w2_bytes = (size_t)OUT * KTOT * sizeof(_Float16);        // ~4.7 MB
  size_t jf_off = (w2_bytes + 255) & ~(size_t)255;
  _Float16* Jf = (_Float16*)((char*)d_ws + jf_off);               // ~37.7 MB

  fuse_wc<<<(OUT * IN + 255) / 256, 256, 0, stream>>>(w, c, W2);
  jacobi_k<<<(BATCH * IN + 255) / 256, 256, 0, stream>>>(x, Jf);

  int waveTiles = (BATCH / 32) * (OUT / 32); // 2048
  gemm_wmma<<<waveTiles / 8, 256, 0, stream>>>(Jf, W2, out);
}
